// MultiheadSelfAttention_5153960755735
// MI455X (gfx1250) — compile-verified, matrix operand order checked by reading, not run on hardware
//
#include <hip/hip_runtime.h>
#include <hip/hip_bf16.h>
#include <math.h>
#include <type_traits>

// ---------------------------------------------------------------------------
// MHA w/ RoPE for gfx1250 (MI455X). All GEMM work through v_wmma_f32_16x16x32_bf16.
// B=2, S=2048, D_MODEL=1024, H=16, Dh=64.
// ---------------------------------------------------------------------------

typedef __attribute__((ext_vector_type(16))) __bf16 v16bf;
typedef __attribute__((ext_vector_type(8)))  __bf16 v8bf;
typedef __attribute__((ext_vector_type(4)))  __bf16 v4bf;
typedef __attribute__((ext_vector_type(8)))  float  v8f;

constexpr int DM   = 1024;   // d_model
constexpr int SEQ  = 2048;   // sequence length
constexpr int NB   = 2;      // batch
constexpr int NH   = 16;     // heads
constexpr int DH   = 64;     // head dim
constexpr int MTOT = NB * SEQ; // 4096 rows

// score scale folded into Q at projection time: 1/sqrt(Dh) * log2(e)
constexpr float SCORE_SCALE = 0.125f * 1.4426950408889634f;

// ---- WMMA wrapper (16x16x32 bf16 -> f32 accumulate) -----------------------
__device__ __forceinline__ v8f wmma_bf16(v16bf a, v16bf b, v8f c) {
  // (neg_a, A, neg_b, B, c_mod, C, reuse_a, reuse_b)
  return __builtin_amdgcn_wmma_f32_16x16x32_bf16(false, a, false, b, (short)0, c, false, false);
}

// ---- fast exp2 ------------------------------------------------------------
__device__ __forceinline__ float fast_exp2(float x) {
#if __has_builtin(__builtin_amdgcn_exp2f)
  return __builtin_amdgcn_exp2f(x);   // v_exp_f32
#else
  return exp2f(x);
#endif
}

// ---- xor-mask lane exchange within 16-lane halves -------------------------
// v_permlane16_b32 (pure VALU) instead of ds_bpermute (LDS unit).
template <int M>
__device__ __forceinline__ float lanexor16(float v) {
#if __has_builtin(__builtin_amdgcn_permlane16)
  constexpr unsigned s0 = ((0u ^ M) << 0)  | ((1u ^ M) << 4)  | ((2u ^ M) << 8)  | ((3u ^ M) << 12) |
                          ((4u ^ M) << 16) | ((5u ^ M) << 20) | ((6u ^ M) << 24) | ((7u ^ M) << 28);
  constexpr unsigned s1 = (((8u ^ M) & 15u) << 0)  | (((9u ^ M) & 15u) << 4)  |
                          (((10u ^ M) & 15u) << 8) | (((11u ^ M) & 15u) << 12) |
                          (((12u ^ M) & 15u) << 16)| (((13u ^ M) & 15u) << 20) |
                          (((14u ^ M) & 15u) << 24)| (((15u ^ M) & 15u) << 28);
  int r = __builtin_amdgcn_permlane16(__float_as_int(v), __float_as_int(v),
                                      (int)s0, (int)s1, false, false);
  return __int_as_float(r);
#else
  return __shfl_xor(v, M, 32);
#endif
}

__device__ __forceinline__ float redmax16(float v) {
  v = fmaxf(v, lanexor16<1>(v));
  v = fmaxf(v, lanexor16<2>(v));
  v = fmaxf(v, lanexor16<4>(v));
  v = fmaxf(v, lanexor16<8>(v));
  return v;
}
__device__ __forceinline__ float redsum16(float v) {
  v += lanexor16<1>(v);
  v += lanexor16<2>(v);
  v += lanexor16<4>(v);
  v += lanexor16<8>(v);
  return v;
}

// ---- 16-bit operand fragment loader ---------------------------------------
// Lane layout (A: row = lane&15; B: col = lane&15; hf = lane>>4):
//   element e (0..15) maps to k = (e&7) + 16*(e>>3) + 8*hf
// => two contiguous 16B chunks per lane: [k0 = 8*hf .. +7] and [16+8*hf .. +7]
__device__ __forceinline__ v16bf load_frag(const __bf16* base, int hf) {
  union { v16bf v; v8bf h[2]; } u;
  u.h[0] = *(const v8bf*)(base + 8 * hf);
  u.h[1] = *(const v8bf*)(base + 16 + 8 * hf);
  return u.v;
}

// ---------------------------------------------------------------------------
// f32 -> bf16 convert (vectorized x4)
// ---------------------------------------------------------------------------
__global__ void cvt_bf16_kernel(const float* __restrict__ in, __bf16* __restrict__ out, int n4) {
  int i = blockIdx.x * blockDim.x + threadIdx.x;
  if (i < n4) {
    float4 f = *(const float4*)(in + 4 * (size_t)i);
    union { v4bf v; __bf16 e[4]; } p;
    p.e[0] = (__bf16)f.x; p.e[1] = (__bf16)f.y;
    p.e[2] = (__bf16)f.z; p.e[3] = (__bf16)f.w;
    *(v4bf*)(out + 4 * (size_t)i) = p.v;
  }
}

// ---------------------------------------------------------------------------
// GEMM: Y[m,n] = sum_k X[m,k] * W[n,k]   (M=4096, N=1024, K=1024)
// Block = 256 thr = 8 waves; block tile 128(M) x 64(N); wave tile 16(M) x 64(N).
// MODE 0: f32 row-major [M,N] store (final projection -> d_out)
// MODE 1: RoPE epilogue * postScale, bf16 store at [b,h,s,dh]  (Q and K)
// MODE 2: bf16 store transposed at [b,h,dh,s]                  (V)
// ---------------------------------------------------------------------------
template <int MODE>
__global__ __launch_bounds__(256) void gemm16_kernel(const __bf16* __restrict__ X,
                                                     const __bf16* __restrict__ W,
                                                     void* __restrict__ outp,
                                                     float postScale) {
  const int lane = threadIdx.x & 31;
  const int wv   = threadIdx.x >> 5;
  const int ln   = lane & 15;
  const int hf   = lane >> 4;
  const int mWave  = blockIdx.x * 128 + wv * 16;
  const int nBlock = blockIdx.y * 64;

  v8f acc[4] = {};
  const __bf16* arow = X + (size_t)(mWave + ln) * DM;

  #pragma unroll 1
  for (int kk = 0; kk < DM; kk += 32) {
    v16bf a = load_frag(arow + kk, hf);
    #pragma unroll
    for (int t = 0; t < 4; ++t) {
      const int n = nBlock + t * 16 + ln;
      v16bf b = load_frag(W + (size_t)n * DM + kk, hf);
      acc[t] = wmma_bf16(a, b, acc[t]);
    }
  }

  if constexpr (MODE == 0) {
    float* out = (float*)outp;
    #pragma unroll
    for (int t = 0; t < 4; ++t) {
      const int n = nBlock + t * 16 + ln;
      #pragma unroll
      for (int r = 0; r < 8; ++r) {
        const int mr = mWave + r + hf * 8;
        out[(size_t)mr * DM + n] = acc[t][r];
      }
    }
  } else if constexpr (MODE == 1) {
    // RoPE: pair (2i, 2i+1) within each head. Adjacent columns = adjacent lanes.
    __bf16* out = (__bf16*)outp;
    #pragma unroll
    for (int t = 0; t < 4; ++t) {
      const int n  = nBlock + t * 16 + ln;
      const int h  = n >> 6;
      const int dh = n & 63;
      const int pair = dh >> 1;
      const bool odd = (dh & 1) != 0;
      // inv_freq = 10000^(-2i/64) = 2^(-2i/64 * log2(10000))
      const float inv_freq = fast_exp2(-(float)(2 * pair) * (13.287712379549449f / 64.f));
      #pragma unroll
      for (int r = 0; r < 8; ++r) {
        const int mr = mWave + r + hf * 8;
        const int b  = mr >> 11;         // / SEQ
        const int s  = mr & (SEQ - 1);   // % SEQ
        float sn, cs;
        __sincosf((float)s * inv_freq, &sn, &cs);
        const float mine  = acc[t][r];
        const float other = lanexor16<1>(mine); // partner column in pair
        float res = odd ? (other * sn + mine * cs)   // x1*sin + x2*cos
                        : (mine * cs - other * sn);  // x1*cos - x2*sin
        res *= postScale;                            // score scale folded into Q
        out[(((size_t)(b * NH + h) * SEQ + s) << 6) + dh] = (__bf16)res;
      }
    }
  } else { // MODE == 2 : V transposed [b,h,dh,s]; 8 consecutive s per lane -> 16B store
    __bf16* out = (__bf16*)outp;
    const int mr0 = mWave + hf * 8;
    const int b   = mr0 >> 11;
    const int s0  = mr0 & (SEQ - 1);
    #pragma unroll
    for (int t = 0; t < 4; ++t) {
      const int n  = nBlock + t * 16 + ln;
      const int h  = n >> 6;
      const int dh = n & 63;
      union { v8bf v; __bf16 e[8]; } pk;
      #pragma unroll
      for (int r = 0; r < 8; ++r) pk.e[r] = (__bf16)acc[t][r];
      *(v8bf*)(out + (size_t)((b * NH + h) * DH + dh) * SEQ + s0) = pk.v;
    }
  }
}

// ---------------------------------------------------------------------------
// Causal flash attention. Grid (SEQ/128, NB*NH), 256 thr = 8 waves.
// Wave owns 16 query rows; streams 32-key tiles; online softmax in log2 domain
// (score scale is pre-folded into Q). Exactly ONE tile per wave needs the
// causal mask: jMax = (qBase+15) & ~31; all tiles below it are fully valid.
// Q,K in [b,h,s,dh] bf16 ; V in [b,h,dh,s] bf16 ; out [b,s,h*64+dh] bf16.
// ---------------------------------------------------------------------------
__global__ __launch_bounds__(256) void attn_kernel(const __bf16* __restrict__ Q,
                                                   const __bf16* __restrict__ K,
                                                   const __bf16* __restrict__ Vt,
                                                   __bf16* __restrict__ O) {
  __shared__ __bf16 pbuf[8][16][32]; // per-wave P tile (q x key), 8 KB total

  const int lane = threadIdx.x & 31;
  const int wv   = threadIdx.x >> 5;
  const int ln   = lane & 15;
  const int hf   = lane >> 4;
  const int bh   = blockIdx.y;                    // b*NH + h
  const int qBase = blockIdx.x * 128 + wv * 16;
  const size_t baseQK = (size_t)bh * SEQ * DH;
  const size_t baseV  = (size_t)bh * DH * SEQ;

  // Q fragments: 16 rows x 64 dh -> two A-frags (k = dh 0..31, 32..63)
  const __bf16* qrow = Q + baseQK + (size_t)(qBase + ln) * DH;
  const v16bf qa0 = load_frag(qrow, hf);
  const v16bf qa1 = load_frag(qrow + 32, hf);

  v8f o[4] = {};
  float Mrow[8], Lrow[8];
  #pragma unroll
  for (int r = 0; r < 8; ++r) { Mrow[r] = -INFINITY; Lrow[r] = 0.f; }

  auto do_tile = [&](int jB, auto mask_c) {
    constexpr bool MASK = decltype(mask_c)::value;
    // ---- scores: S(16x32) = Q(16x64) . K^T(64x32), two 16x16 C tiles ----
    v8f s0 = {}, s1 = {};
    {
      const __bf16* k0 = K + baseQK + (size_t)(jB + ln) * DH;
      const __bf16* k1 = k0 + 16 * DH;
      s0 = wmma_bf16(qa0, load_frag(k0, hf), s0);
      s0 = wmma_bf16(qa1, load_frag(k0 + 32, hf), s0);
      s1 = wmma_bf16(qa0, load_frag(k1, hf), s1);
      s1 = wmma_bf16(qa1, load_frag(k1 + 32, hf), s1);
    }
    // ---- online softmax, log2 domain (rows spread across 16 lanes/half) ----
    #pragma unroll
    for (int r = 0; r < 8; ++r) {
      float a0 = s0[r];
      float a1 = s1[r];
      if constexpr (MASK) {
        const int row = qBase + r + hf * 8;
        if (jB + ln > row)      a0 = -INFINITY;
        if (jB + 16 + ln > row) a1 = -INFINITY;
      }
      const float mloc = redmax16(fmaxf(a0, a1));
      const float mnew = fmaxf(Mrow[r], mloc);
      const float e0 = fast_exp2(a0 - mnew);       // exp2 == exp in log2 domain
      const float e1 = fast_exp2(a1 - mnew);
      const float scale = fast_exp2(Mrow[r] - mnew);
      Lrow[r] = Lrow[r] * scale + redsum16(e0 + e1);
      Mrow[r] = mnew;
      #pragma unroll
      for (int c = 0; c < 4; ++c) o[c][r] = o[c][r] * scale;
      // stage P as bf16 in LDS at [q_row][key] for A-fragment reload
      pbuf[wv][r + hf * 8][ln]      = (__bf16)e0;
      pbuf[wv][r + hf * 8][ln + 16] = (__bf16)e1;
    }
    // DS ops are in-order within a wave; fence the compiler + drain counter
    asm volatile("s_wait_dscnt 0" ::: "memory");
    // ---- O += P(16x32) . V(32x64): 4 WMMAs (n tiles of 16 dh columns) ----
    const v16bf pa = load_frag(&pbuf[wv][ln][0], hf);
    #pragma unroll
    for (int c = 0; c < 4; ++c) {
      const __bf16* vb = Vt + baseV + (size_t)(c * 16 + ln) * SEQ + jB;
      o[c] = wmma_bf16(pa, load_frag(vb, hf), o[c]);
    }
  };

  const int jMax = (qBase + 15) & ~31;          // the single masked tile
  for (int jB = 0; jB < jMax; jB += 32)
    do_tile(jB, std::false_type{});             // steady state: no mask code
  do_tile(jMax, std::true_type{});              // diagonal tile: masked

  // ---- normalize + store to [b, s, h*64 + dh] as bf16 ----
  const int b = bh >> 4, h = bh & 15;
  #pragma unroll
  for (int c = 0; c < 4; ++c) {
    #pragma unroll
    for (int r = 0; r < 8; ++r) {
      const int s = qBase + r + hf * 8;
      const float val = o[c][r] / Lrow[r];
      O[(size_t)(b * SEQ + s) * DM + h * DH + c * 16 + ln] = (__bf16)val;
    }
  }
}

// ---------------------------------------------------------------------------
// Launcher. Workspace layout (bytes):
//   [0,8M)   Xbf      [8M,10M) Wq  [10M,12M) Wk  [12M,14M) Wv  [14M,16M) Wo
//   [16M,24M) Q[b,h,s,dh]  [24M,32M) K[b,h,s,dh]  [32M,40M) Vt[b,h,dh,s]
//   [40M,48M) attn out [b,s,e]                     (all bf16)
// ---------------------------------------------------------------------------
extern "C" void kernel_launch(void* const* d_in, const int* in_sizes, int n_in,
                              void* d_out, int out_size, void* d_ws, size_t ws_size,
                              hipStream_t stream) {
  const float* Xf  = (const float*)d_in[0];
  const float* Wqf = (const float*)d_in[1];
  const float* Wkf = (const float*)d_in[2];
  const float* Wvf = (const float*)d_in[3];
  const float* Wof = (const float*)d_in[4];

  char* ws = (char*)d_ws;
  const size_t MB = (size_t)1 << 20;
  __bf16* Xb  = (__bf16*)(ws + 0 * MB);
  __bf16* Wqb = (__bf16*)(ws + 8 * MB);
  __bf16* Wkb = (__bf16*)(ws + 10 * MB);
  __bf16* Wvb = (__bf16*)(ws + 12 * MB);
  __bf16* Wob = (__bf16*)(ws + 14 * MB);
  __bf16* Qb  = (__bf16*)(ws + 16 * MB);
  __bf16* Kb  = (__bf16*)(ws + 24 * MB);
  __bf16* Vt  = (__bf16*)(ws + 32 * MB);
  __bf16* AO  = (__bf16*)(ws + 40 * MB);

  // 1) convert inputs to bf16
  const int nX4 = MTOT * DM / 4;   // 1,048,576
  const int nW4 = DM * DM / 4;     //   262,144
  cvt_bf16_kernel<<<(nX4 + 255) / 256, 256, 0, stream>>>(Xf,  Xb,  nX4);
  cvt_bf16_kernel<<<(nW4 + 255) / 256, 256, 0, stream>>>(Wqf, Wqb, nW4);
  cvt_bf16_kernel<<<(nW4 + 255) / 256, 256, 0, stream>>>(Wkf, Wkb, nW4);
  cvt_bf16_kernel<<<(nW4 + 255) / 256, 256, 0, stream>>>(Wvf, Wvb, nW4);
  cvt_bf16_kernel<<<(nW4 + 255) / 256, 256, 0, stream>>>(Wof, Wob, nW4);

  // 2) QKV projections (RoPE + score-scale fused into Q; V stored transposed)
  dim3 gGemm(MTOT / 128, DM / 64); // (32, 16)
  gemm16_kernel<1><<<gGemm, 256, 0, stream>>>(Xb, Wqb, Qb, SCORE_SCALE);
  gemm16_kernel<1><<<gGemm, 256, 0, stream>>>(Xb, Wkb, Kb, 1.0f);
  gemm16_kernel<2><<<gGemm, 256, 0, stream>>>(Xb, Wvb, Vt, 1.0f);

  // 3) causal flash attention
  attn_kernel<<<dim3(SEQ / 128, NB * NH), 256, 0, stream>>>(Qb, Kb, Vt, AO);

  // 4) output projection -> f32 d_out
  gemm16_kernel<0><<<gGemm, 256, 0, stream>>>(AO, Wob, (float*)d_out, 1.0f);
}
